// FreqConv1d_32650341384313
// MI455X (gfx1250) — compile-verified, matrix operand order checked by reading, not run on hardware
//
#include <hip/hip_runtime.h>
#include <cstdint>
#include <cstddef>

// ---------------------------------------------------------------------------
// FreqConv1d on MI455X (gfx1250): all FFT stages expressed as bf16 WMMA GEMMs,
// with CDNA5 async global->LDS DMA (ASYNCcnt) double-buffered staging.
//
//  x[16,64,4096] --cvt--> Xbf[1024,4096] (bf16)
//  w[64,64,64]   --cvt--> Wbf[4096,64]   (bf16)
//  GEMM1: Xf[1024,2048]  = Xbf · DxT^T      (forward rDFT, re/im interleaved)
//  GEMM2: Kf[4096,2048]  = Wbf · WkT^T      (kernel rDFT w/ 4032-sample shift)
//  EINSUM: Cbf[1024,2048] = per-f complex contraction over Cin (LDS tiled)
//  GEMM3: out[1024,2046] = Cbf · iDT^T + bias   (weighted inverse rDFT /2046)
// ---------------------------------------------------------------------------

typedef unsigned short u16;
typedef __attribute__((ext_vector_type(16))) __bf16 v16bf;
typedef __attribute__((ext_vector_type(8)))  float  v8f;

static constexpr int NB    = 16;    // batch
static constexpr int CIN   = 64;
static constexpr int COUT  = 64;
static constexpr int TT    = 4096;  // input time length
static constexpr int KW    = 64;    // kernel taps
static constexpr int FQ    = 1024;  // kept frequency bins (F//2)
static constexpr int FJ    = 2048;  // re/im interleaved bins
static constexpr int TOUT  = 2046;  // irfft output length 2*(FQ-1)
static constexpr int TOUTP = 2048;  // padded N for GEMM3

__device__ __forceinline__ u16 f2bf(float f) {
  union { float f; uint32_t u; } v; v.f = f;
  uint32_t r = v.u + 0x7FFFu + ((v.u >> 16) & 1u);   // round-to-nearest-even
  return (u16)(r >> 16);
}

// CDNA5 async global->LDS DMA helpers (tracked by ASYNCcnt).
__device__ __forceinline__ void async_lds_b128(uint32_t ldsoff, const void* g) {
  asm volatile("global_load_async_to_lds_b128 %0, %1, off"
               :: "v"(ldsoff), "v"(g) : "memory");
}
__device__ __forceinline__ void async_lds_b128_o16(uint32_t ldsoff, const void* g) {
  asm volatile("global_load_async_to_lds_b128 %0, %1, off offset:16"
               :: "v"(ldsoff), "v"(g) : "memory");
}
__device__ __forceinline__ void async_lds_b64(uint32_t ldsoff, const void* g) {
  asm volatile("global_load_async_to_lds_b64 %0, %1, off"
               :: "v"(ldsoff), "v"(g) : "memory");
}
__device__ __forceinline__ void wait_async0() {
  asm volatile("s_wait_asynccnt 0" ::: "memory");
}
__device__ __forceinline__ uint32_t lds_off(const void* p) {
  return (uint32_t)(uintptr_t)p;   // generic LDS ptr: low 32 bits = DS offset
}

// ------------------------- basis-table builders ----------------------------

// DxT[fj][t], fj-major, stride TT.  2f -> cos(2*pi*f*t/4096), 2f+1 -> -sin(..)
__global__ void build_dxT(u16* __restrict__ DxT) {
  int i = blockIdx.x * 256 + threadIdx.x;
  if (i >= FJ * TT) return;
  int t = i & (TT - 1);
  int fj = i >> 12;
  int f = fj >> 1, im = fj & 1;
  int p = (int)(((long long)f * t) & (TT - 1));      // exact phase mod 4096
  float th = 6.283185307179586f * (float)p / (float)TT;
  float s, c;
  __sincosf(th, &s, &c);
  DxT[i] = f2bf(im ? -s : c);
}

// WkT[fj][k], stride KW. phase uses sample position (4096-64)+k.
__global__ void build_wkT(u16* __restrict__ WkT) {
  int i = blockIdx.x * 256 + threadIdx.x;
  if (i >= FJ * KW) return;
  int k = i & (KW - 1);
  int fj = i >> 6;
  int f = fj >> 1, im = fj & 1;
  int p = (int)(((long long)f * (TT - KW + k)) & (TT - 1));
  float th = 6.283185307179586f * (float)p / (float)TT;
  float s, c;
  __sincosf(th, &s, &c);
  WkT[i] = f2bf(im ? -s : c);
}

// iDT[t][fj], stride FJ, t padded to 2048 (rows >= 2046 are zero).
// 2f -> a_f*cos(2*pi*f*t/2046)/2046 ; 2f+1 -> -a_f*sin(..)/2046, a_f in {1,2}.
__global__ void build_idT(u16* __restrict__ iDT) {
  int i = blockIdx.x * 256 + threadIdx.x;
  if (i >= TOUTP * FJ) return;
  int fj = i & (FJ - 1);
  int t = i >> 11;
  if (t >= TOUT) { iDT[i] = 0; return; }
  int f = fj >> 1, im = fj & 1;
  int p = (int)(((long long)f * t) % TOUT);
  float th = 6.283185307179586f * (float)p / (float)TOUT;
  float s, c;
  __sincosf(th, &s, &c);
  float a = ((f == 0) || (f == FQ - 1)) ? 1.0f : 2.0f;
  a *= (1.0f / (float)TOUT);
  iDT[i] = f2bf(im ? (-a * s) : (a * c));
}

__global__ void cvt_f32_bf16(const float* __restrict__ src, u16* __restrict__ dst, int n) {
  int i = blockIdx.x * 256 + threadIdx.x;
  if (i < n) dst[i] = f2bf(src[i]);
}

// --------------------------- WMMA GEMM core --------------------------------
// C[M x Nvalid] = A[M x Kt] * Bt[Npad x Kt]^T  (A,Bt bf16 row-major; C f32)
// Block tile 128x128, BK=32, 8 wave32s (2x4), each wave 64x32 (8 accumulators).
// Double-buffered LDS, filled by global_load_async_to_lds_b128 (ASYNCcnt).

__device__ __forceinline__ v16bf ld_frag(const u16* __restrict__ lds,
                                         int row, int khalf) {
  union { v16bf v; uint4 q[2]; } u;
  const uint4* p = (const uint4*)(lds + row * 40);   // 40 u16 = 80 B stride
  u.q[0] = p[khalf];        // K = khalf*8 + (0..7)
  u.q[1] = p[2 + khalf];    // K = 16 + khalf*8 + (0..7)
  return u.v;
}

template <bool BIAS>
__global__ __launch_bounds__(256)
void gemm_bf16_wmma(const u16* __restrict__ A, const u16* __restrict__ Bt,
                    float* __restrict__ C, int M, int Npad, int Kt,
                    int ldc, int nvalid, const float* __restrict__ bias) {
  __shared__ u16 lA[2][128 * 40];   // 10 KB per buffer
  __shared__ u16 lB[2][128 * 40];

  const int tid   = threadIdx.x;
  const int lane  = tid & 31;
  const int wid   = tid >> 5;
  const int wm    = (wid & 1) * 64;    // wave M origin in tile
  const int wn    = (wid >> 1) * 32;   // wave N origin in tile
  const int bm    = blockIdx.y * 128;
  const int bn    = blockIdx.x * 128;
  const int row   = tid >> 1;          // 0..127 staging row
  const int seg   = (tid & 1) * 16;    // 0 | 16 staging col
  const int khalf = lane >> 4;
  const int mlo   = lane & 15;

  const u16* gA = A  + (size_t)(bm + row) * Kt + seg;   // + k0 per stage
  const u16* gB = Bt + (size_t)(bn + row) * Kt + seg;
  const uint32_t sA0 = lds_off(&lA[0][row * 40 + seg]);
  const uint32_t sB0 = lds_off(&lB[0][row * 40 + seg]);
  const uint32_t bstride = (uint32_t)(128 * 40 * sizeof(u16));  // 10240 B

  // stage(k0) into buffer b: each thread DMAs 2x16B per matrix
  auto stage = [&](int b, int k0) {
    uint32_t sa = sA0 + (uint32_t)b * bstride;
    uint32_t sb = sB0 + (uint32_t)b * bstride;
    async_lds_b128(sa, gA + k0);
    async_lds_b128_o16(sa, gA + k0);     // offset applies to LDS and global
    async_lds_b128(sb, gB + k0);
    async_lds_b128_o16(sb, gB + k0);
  };

  v8f acc[8] = {};
  stage(0, 0);                            // prologue
  int buf = 0;

  for (int k0 = 0; k0 < Kt; k0 += 32) {
    wait_async0();          // my DMAs into `buf` landed
    __syncthreads();        // everyone's landed; everyone done reading buf^1
    if (k0 + 32 < Kt) stage(buf ^ 1, k0 + 32);   // overlap next tile DMA

    const u16* cA = lA[buf];
    const u16* cB = lB[buf];
    v16bf afr[4], bfr[2];
#pragma unroll
    for (int mt = 0; mt < 4; ++mt)
      afr[mt] = ld_frag(cA, wm + mt * 16 + mlo, khalf);
#pragma unroll
    for (int nt = 0; nt < 2; ++nt)
      bfr[nt] = ld_frag(cB, wn + nt * 16 + mlo, khalf);

#pragma unroll
    for (int mt = 0; mt < 4; ++mt)
#pragma unroll
      for (int nt = 0; nt < 2; ++nt)
        acc[mt * 2 + nt] = __builtin_amdgcn_wmma_f32_16x16x32_bf16(
            false, afr[mt], false, bfr[nt], (short)0, acc[mt * 2 + nt],
            false, false);
    buf ^= 1;
  }

  // ---- epilogue: 16x16 f32 C layout (lane<16: M=r, lane>=16: M=r+8) ----
  const int mhi = (lane < 16) ? 0 : 8;
#pragma unroll
  for (int mt = 0; mt < 4; ++mt) {
#pragma unroll
    for (int nt = 0; nt < 2; ++nt) {
      v8f a = acc[mt * 2 + nt];
#pragma unroll
      for (int r = 0; r < 8; ++r) {
        int gm = bm + wm + mt * 16 + mhi + r;
        int gn = bn + wn + nt * 16 + mlo;
        if (gn < nvalid && gm < M) {
          float v = a[r];
          if (BIAS) v += bias[gm & 63];          // row = b*64 + o
          C[(size_t)gm * ldc + gn] = v;
        }
      }
    }
  }
}

// ---------------- per-frequency complex contraction over Cin ---------------
// One workgroup per frequency f. LDS: X[16][64]{re,im} + K[64][64]{re,im},
// filled with global_load_async_to_lds_b64 (re/im pairs are contiguous).
__global__ __launch_bounds__(256)
void einsum_freq(const float* __restrict__ Xf, const float* __restrict__ Kf,
                 u16* __restrict__ Cbf) {
  __shared__ float Xs[NB * CIN * 2];     // 8 KB
  __shared__ float Ks[COUT * CIN * 2];   // 32 KB
  const int f = blockIdx.x;              // 0..1023
  const int tid = threadIdx.x;
  const uint32_t xsb = lds_off(Xs);
  const uint32_t ksb = lds_off(Ks);

  for (int i = tid; i < NB * CIN; i += 256)          // i = b*64 + c
    async_lds_b64(xsb + (uint32_t)i * 8u, Xf + (size_t)i * FJ + 2 * f);
  for (int i = tid; i < COUT * CIN; i += 256)        // i = o*64 + c
    async_lds_b64(ksb + (uint32_t)i * 8u, Kf + (size_t)i * FJ + 2 * f);
  wait_async0();
  __syncthreads();

  for (int idx = tid; idx < NB * COUT; idx += 256) {  // idx = b*64 + o
    const int b = idx >> 6, o = idx & 63;
    const float* xp = &Xs[b * CIN * 2];
    const float* kp = &Ks[o * CIN * 2];
    float cr = 0.f, ci = 0.f;
#pragma unroll 8
    for (int c = 0; c < CIN; ++c) {
      float xr = xp[2 * c], xi = xp[2 * c + 1];
      float kr = kp[2 * c], ki = kp[2 * c + 1];
      cr += xr * kr - xi * ki;
      ci += xr * ki + xi * kr;
    }
    Cbf[(size_t)idx * FJ + 2 * f]     = f2bf(cr);
    Cbf[(size_t)idx * FJ + 2 * f + 1] = f2bf(ci);
  }
}

// ------------------------------ launcher -----------------------------------

extern "C" void kernel_launch(void* const* d_in, const int* in_sizes, int n_in,
                              void* d_out, int out_size, void* d_ws, size_t ws_size,
                              hipStream_t stream) {
  const float* x    = (const float*)d_in[0];  // [16,64,4096]
  const float* w    = (const float*)d_in[1];  // [64,64,64]
  const float* bias = (const float*)d_in[2];  // [64]
  float* out = (float*)d_out;                 // [16,64,2046]

  // workspace carve-up (256B aligned), ~77 MB total
  char* p = (char*)d_ws;
  auto take = [&](size_t bytes) {
    char* r = p;
    p += (bytes + 255) & ~(size_t)255;
    return r;
  };
  u16*   DxT = (u16*)  take((size_t)FJ * TT * 2);       // 16.8 MB
  u16*   WkT = (u16*)  take((size_t)FJ * KW * 2);       // 256 KB
  u16*   iDT = (u16*)  take((size_t)TOUTP * FJ * 2);    // 8.4 MB
  u16*   Xbf = (u16*)  take((size_t)NB * CIN * TT * 2); // 8.4 MB
  u16*   Wbf = (u16*)  take((size_t)COUT * CIN * KW * 2); // 512 KB
  float* Xf  = (float*)take((size_t)NB * CIN * FJ * 4); // 8.4 MB
  float* Kf  = (float*)take((size_t)COUT * CIN * FJ * 4); // 33.6 MB
  u16*   Cbf = (u16*)  take((size_t)NB * COUT * FJ * 2);  // 4.2 MB

  // 1) basis tables + input precision conversion
  build_dxT<<<(FJ * TT) / 256, 256, 0, stream>>>(DxT);
  build_wkT<<<(FJ * KW) / 256, 256, 0, stream>>>(WkT);
  build_idT<<<(TOUTP * FJ) / 256, 256, 0, stream>>>(iDT);
  cvt_f32_bf16<<<(NB * CIN * TT) / 256, 256, 0, stream>>>(x, Xbf, NB * CIN * TT);
  cvt_f32_bf16<<<(COUT * CIN * KW) / 256, 256, 0, stream>>>(w, Wbf, COUT * CIN * KW);

  // 2) forward rDFT of x:   Xf[1024,2048] = Xbf[1024,4096] . DxT^T
  gemm_bf16_wmma<false><<<dim3(FJ / 128, (NB * CIN) / 128), 256, 0, stream>>>(
      Xbf, DxT, Xf, NB * CIN, FJ, TT, FJ, FJ, nullptr);

  // 3) kernel rDFT:         Kf[4096,2048] = Wbf[4096,64] . WkT^T
  gemm_bf16_wmma<false><<<dim3(FJ / 128, (COUT * CIN) / 128), 256, 0, stream>>>(
      Wbf, WkT, Kf, COUT * CIN, FJ, KW, FJ, FJ, nullptr);

  // 4) per-frequency complex contraction over Cin
  einsum_freq<<<FQ, 256, 0, stream>>>(Xf, Kf, Cbf);

  // 5) weighted inverse rDFT + bias: out[1024,2046] = Cbf[1024,2048] . iDT^T
  gemm_bf16_wmma<true><<<dim3(TOUTP / 128, (NB * COUT) / 128), 256, 0, stream>>>(
      Cbf, iDT, out, NB * COUT, TOUTP, FJ, TOUT, TOUT, bias);
}
